// W8A8Linear_84782654423420
// MI455X (gfx1250) — 8-bit matrix operand order repaired and checked by reading, compiled, not run on hardware
//
#include <hip/hip_runtime.h>
#include <hip/hip_bf16.h>
#include <stdint.h>

#define M_DIM 16384
#define K_DIM 4096
#define N_DIM 4096

typedef __attribute__((ext_vector_type(8))) int      v8i;
typedef __attribute__((ext_vector_type(8))) _Float16 v8h;

__global__ __launch_bounds__(256) void w8a8_quant_kernel(
    const _Float16* __restrict__ x, int8_t* __restrict__ Aq,
    float* __restrict__ xscale) {
  const int row = blockIdx.x;
  const int tid = threadIdx.x;
  const size_t base = (size_t)row * K_DIM + (size_t)tid * 16;

  v8h h0 = *(const v8h*)(x + base);
  v8h h1 = *(const v8h*)(x + base + 8);

  float f[16];
  float mx = 0.0f;
#pragma unroll
  for (int i = 0; i < 8; ++i) { f[i]     = (float)h0[i]; mx = fmaxf(mx, fabsf(f[i])); }
#pragma unroll
  for (int i = 0; i < 8; ++i) { f[8 + i] = (float)h1[i]; mx = fmaxf(mx, fabsf(f[8 + i])); }

#pragma unroll
  for (int off = 16; off >= 1; off >>= 1)
    mx = fmaxf(mx, __shfl_xor(mx, off, 32));

  __shared__ float smax[8];
  __shared__ float sscale;
  if ((tid & 31) == 0) smax[tid >> 5] = mx;
  __syncthreads();
  if (tid == 0) {
    float m = smax[0];
#pragma unroll
    for (int i = 1; i < 8; ++i) m = fmaxf(m, smax[i]);
    m = fmaxf(m, 1e-8f);
    float sc = m / 127.0f;
    sscale = sc;
    xscale[row] = sc;
  }
  __syncthreads();

  const float inv = 1.0f / sscale;
  int dw[4];
#pragma unroll
  for (int d = 0; d < 4; ++d) {
    unsigned pack = 0;
#pragma unroll
    for (int b = 0; b < 4; ++b) {
      int q = (int)rintf(f[d * 4 + b] * inv);
      q = q < -128 ? -128 : (q > 127 ? 127 : q);
      pack |= ((unsigned)(q & 0xff)) << (8 * b);
    }
    dw[d] = (int)pack;
  }
  *(int4*)(Aq + base) = make_int4(dw[0], dw[1], dw[2], dw[3]);
}

__global__ __launch_bounds__(256) void w8a8_repack_kernel(
    const int8_t* __restrict__ w, int32_t* __restrict__ Wp) {
  const int kb64 = K_DIM / 64;
  const int total = (N_DIM / 16) * kb64 * 32 * 8;
  int id = blockIdx.x * 256 + threadIdx.x;
  if (id >= total) return;

  const int v = id & 7;
  const int l = (id >> 3) & 31;
  const int tile = id >> 8;
  const int kb = tile % kb64;
  const int nb = tile / kb64;

  const int n = nb * 16 + (l & 15);
  const int kbase = kb * 64 + ((l >> 4) * 8) + ((v & 1) * 4) + ((v >> 1) * 16);

  unsigned pack = 0;
#pragma unroll
  for (int b = 0; b < 4; ++b) {
    int8_t e = w[(size_t)(kbase + b) * N_DIM + n];
    pack |= ((unsigned)(uint8_t)e) << (8 * b);
  }
  Wp[id] = (int)pack;
}

__global__ __launch_bounds__(256) void w8a8_gemm_kernel(
    const int8_t* __restrict__ Aq, const int8_t* __restrict__ Wp,
    const float* __restrict__ xscale, const float* __restrict__ wscale,
    const _Float16* __restrict__ bias, _Float16* __restrict__ out) {
  __shared__ __align__(16) int8_t sA[2][128 * 64];

  const int tid = threadIdx.x;
  const int lane = tid & 31;
  const int w = tid >> 5;
  const int wm = w >> 2;
  const int wn = w & 3;
  const int m0 = blockIdx.y * 128;
  const int n0 = blockIdx.x * 256;

  const int row = tid >> 1;
  const int koff = (tid & 1) * 32;

  v8i acc[4][4] = {};

  const unsigned aoff = (unsigned)((m0 + row) * K_DIM + koff);
  const unsigned lds0 = (unsigned)(uintptr_t)&sA[0][row * 64 + koff];
  const unsigned lds1 = lds0 + 128 * 64;

  asm volatile("global_load_async_to_lds_b128 %0, %1, %2 offset:0"
               :: "v"(lds0), "v"(aoff), "s"(Aq) : "memory");
  asm volatile("global_load_async_to_lds_b128 %0, %1, %2 offset:16"
               :: "v"(lds0), "v"(aoff), "s"(Aq) : "memory");

  const int kb64 = K_DIM / 64;
  const int nbBase = (n0 >> 4) + wn * 4;

#pragma unroll 1
  for (int kt = 0; kt < kb64; ++kt) {
    asm volatile("s_wait_asynccnt 0x0" ::: "memory");
    __syncthreads();
    const int cur = kt & 1;

    if (kt + 1 < kb64) {
      const unsigned a2 = aoff + (unsigned)(kt + 1) * 64;
      const unsigned ldsn = cur ? lds0 : lds1;
      asm volatile("global_load_async_to_lds_b128 %0, %1, %2 offset:0"
                   :: "v"(ldsn), "v"(a2), "s"(Aq) : "memory");
      asm volatile("global_load_async_to_lds_b128 %0, %1, %2 offset:16"
                   :: "v"(ldsn), "v"(a2), "s"(Aq) : "memory");
    }

    v8i bf[4];
#pragma unroll
    for (int jn = 0; jn < 4; ++jn) {
      size_t boff = (((size_t)(nbBase + jn) * kb64 + kt) * 32 + lane) * 32;
      int4 lo = *(const int4*)(Wp + boff);
      int4 hi = *(const int4*)(Wp + boff + 16);
      bf[jn][0] = lo.x; bf[jn][1] = lo.y; bf[jn][2] = lo.z; bf[jn][3] = lo.w;
      bf[jn][4] = hi.x; bf[jn][5] = hi.y; bf[jn][6] = hi.z; bf[jn][7] = hi.w;
    }

    const int sel = (lane >> 4) * 8;
#pragma unroll
    for (int jm = 0; jm < 4; ++jm) {
      const int rbase = wm * 64 + jm * 16 + (lane & 15);
      v8i af;
#pragma unroll
      for (int c = 0; c < 4; ++c) {
        int2 dv = *(const int2*)(&sA[cur][rbase * 64 + sel + c * 16]);
        af[2 * c] = dv.x;
        af[2 * c + 1] = dv.y;
      }
#pragma unroll
      for (int jn = 0; jn < 4; ++jn) {
        acc[jm][jn] = __builtin_amdgcn_wmma_i32_16x16x64_iu8(
            true, af, true, bf[jn], acc[jm][jn], false, false);
      }
    }
  }

#pragma unroll
  for (int jn = 0; jn < 4; ++jn) {
    const int col = n0 + wn * 64 + jn * 16 + (lane & 15);
    const float wsc = wscale[col];
    const _Float16 bz = bias[col];
#pragma unroll
    for (int jm = 0; jm < 4; ++jm) {
      const int rbase = m0 + wm * 64 + jm * 16 + ((lane >> 4) * 8);
#pragma unroll
      for (int e = 0; e < 8; ++e) {
        const int rrow = rbase + e;
        float val = (float)acc[jm][jn][e] * xscale[rrow] * wsc;
        out[(size_t)rrow * N_DIM + col] = (_Float16)val + bz;
      }
    }
  }
}

extern "C" void kernel_launch(void* const* d_in, const int* in_sizes, int n_in,
                              void* d_out, int out_size, void* d_ws, size_t ws_size,
                              hipStream_t stream) {
  const _Float16* x      = (const _Float16*)d_in[0];
  const int8_t*   wq     = (const int8_t*)d_in[1];
  const float*    wscale = (const float*)d_in[2];
  const _Float16* bias   = (const _Float16*)d_in[3];
  _Float16*       out    = (_Float16*)d_out;

  char* ws = (char*)d_ws;
  int8_t* Aq = (int8_t*)ws;
  float*  xs = (float*)(ws + (size_t)M_DIM * K_DIM);
  int8_t* Wp = (int8_t*)(ws + (size_t)M_DIM * K_DIM + (1u << 20));

  w8a8_quant_kernel<<<M_DIM, 256, 0, stream>>>(x, Aq, xs);

  {
    const int total_dwords = (N_DIM / 16) * (K_DIM / 64) * 32 * 8;
    w8a8_repack_kernel<<<(total_dwords + 255) / 256, 256, 0, stream>>>(
        wq, (int32_t*)Wp);
  }

  {
    dim3 grid(N_DIM / 256, M_DIM / 128);
    w8a8_gemm_kernel<<<grid, 256, 0, stream>>>(Aq, Wp, xs, wscale, bias, out);
  }
}
